// DeformableConv2d_43602507989498
// MI455X (gfx1250) — compile-verified, matrix operand order checked by reading, not run on hardware
//
#include <hip/hip_runtime.h>
#include <hip/hip_bf16.h>
#include <math.h>

typedef __attribute__((ext_vector_type(16))) _Float16 v16h;
typedef __attribute__((ext_vector_type(8)))  _Float16 v8h;
typedef __attribute__((ext_vector_type(8)))  float    v8f;
typedef __attribute__((ext_vector_type(4)))  unsigned int v4u;
typedef __attribute__((ext_vector_type(4)))  int      v4i;
typedef __attribute__((ext_vector_type(8)))  int      v8i;

#define HW    4096      // 64*64
#define CIN   256
#define OUTC  256
#define NB    8
#define KPTS  9
#define KC    2304      // CIN*KPTS (GEMM reduction dim)
#define TM    64        // pixel-rows per workgroup tile
#define TN    256       // output channels per workgroup tile (all of them)
#define TK    32        // k-chunk (matches wmma f16 K)
#define LDA   40        // padded LDS row stride in halves (80B, conflict-free)

__device__ __forceinline__ int iclamp(int v, int lo, int hi) {
    return v < lo ? lo : (v > hi ? hi : v);
}

// ---------------------------------------------------------------------------
// Kernel 1: 3x3 convs producing offsets (18 ch) and sigmoid-modulation (9 ch)
// ---------------------------------------------------------------------------
__global__ __launch_bounds__(256)
void dcn_offmod_conv(const float* __restrict__ x,
                     const float* __restrict__ ow, const float* __restrict__ ob,
                     const float* __restrict__ mw, const float* __restrict__ mb,
                     float* __restrict__ offs, float* __restrict__ modp)
{
    int tid = blockIdx.x * 256 + threadIdx.x;
    if (tid >= NB * 27 * HW) return;
    int p  = tid & (HW - 1);
    int ch = (tid >> 12) % 27;
    int b  = tid / (27 * HW);
    int py = p >> 6, px = p & 63;

    const float* w;
    float acc;
    if (ch < 18) { w = ow + (size_t)ch * CIN * 9;        acc = ob[ch];      }
    else         { w = mw + (size_t)(ch - 18) * CIN * 9; acc = mb[ch - 18]; }

    const float* xb = x + (size_t)b * CIN * HW + p;
    for (int c = 0; c < CIN; ++c) {
        const float* xp = xb + c * HW;
        const float* wc = w + c * 9;
        #pragma unroll
        for (int t = 0; t < 9; ++t) {
            const int dy = t / 3 - 1, dx = t % 3 - 1;
            const int yy = py + dy, xx = px + dx;
            if (yy >= 0 && yy < 64 && xx >= 0 && xx < 64)
                acc = fmaf(xp[dy * 64 + dx], wc[t], acc);
        }
    }
    if (ch < 18) offs[((size_t)b * 18 + ch) * HW + p] = acc;
    else         modp[((size_t)b * 9 + (ch - 18)) * HW + p] = 1.f / (1.f + __expf(-acc));
}

// ---------------------------------------------------------------------------
// Kernel 1b: convert weight (O,C,3,3) fp32 -> f16, transposed to [k][o][c]
// so each GEMM chunk's B tile is a contiguous-row strided 2D block (TDM-ready)
// ---------------------------------------------------------------------------
__global__ __launch_bounds__(256)
void dcn_wt_convert(const float* __restrict__ w, _Float16* __restrict__ wt16)
{
    int tid = blockIdx.x * 256 + threadIdx.x;
    if (tid >= KPTS * OUTC * CIN) return;
    const int c = tid & 255;
    const int o = (tid >> 8) & 255;
    const int k = tid >> 16;                 // 0..8
    wt16[tid] = (_Float16)w[(size_t)o * KC + c * 9 + k];
}

// ---------------------------------------------------------------------------
// TDM helper: async-load one 32(o) x 32(c) f16 slab of wt16 into LDS with
// padding (row 64B + 16B pad -> LDS stride 80B). Per CDNA5 ISA D# layout.
// ---------------------------------------------------------------------------
__device__ __forceinline__ void tdm_load_b_slab(unsigned int lds_byte_addr,
                                                unsigned long long gaddr)
{
    const unsigned int w1 = (unsigned int)__builtin_amdgcn_readfirstlane((int)lds_byte_addr);
    const unsigned int w2 = (unsigned int)__builtin_amdgcn_readfirstlane((int)(unsigned int)gaddr);
    const unsigned int w3 = ((unsigned int)__builtin_amdgcn_readfirstlane(
                                 (int)(unsigned int)(gaddr >> 32)) & 0x01FFFFFFu)
                          | (2u << 30);      // D# type = 2 ("image")
    v4u g0 = { 1u /*count=1*/, w1, w2, w3 };
    v8i g1 = {
        (int)((1u << 16)       // data_size = 2 bytes
            | (1u << 20)       // pad_enable
            | (3u << 22)       // pad_interval: 16 DWORDs (64B)
            | (3u << 25)),     // pad_amount:   4 DWORDs (16B)
        (int)(256u << 16),     // tensor_dim0 = 256 (low16 in [63:48])
        (int)(2304u << 16),    // dim0 hi16=0 | tensor_dim1 = 2304 (low16)
        (int)(32u << 16),      // dim1 hi16=0 | tile_dim0 = 32
        32,                    // tile_dim1 = 32 rows (o), tile_dim2 = 0
        256,                   // tensor_dim0_stride = 256 elements (low32)
        0, 0
    };
    v4i z4 = { 0, 0, 0, 0 };
#if defined(__clang_major__) && (__clang_major__ >= 23)
    v8i z8 = { 0, 0, 0, 0, 0, 0, 0, 0 };
    __builtin_amdgcn_tensor_load_to_lds(g0, g1, z4, z4, z8, 0);
#else
    __builtin_amdgcn_tensor_load_to_lds(g0, g1, z4, z4, 0);
#endif
}

// ---------------------------------------------------------------------------
// Kernel 2: fused bilinear-sample + modulate + implicit GEMM via WMMA f16
// out[b,o,p] = bias[o] + sum_{k,c} sample(b,p,k,c)*mod(b,p,k) * W[o,c,k]
// ---------------------------------------------------------------------------
__global__ __launch_bounds__(256)
void dcn_wmma_gemm(const float* __restrict__ x,
                   const _Float16* __restrict__ wt16,   // [k][o][c] f16
                   const float* __restrict__ bias,
                   const float* __restrict__ offs,
                   const float* __restrict__ modp,
                   float* __restrict__ out)
{
    __shared__ __align__(16) float4   meta_w[TM * KPTS];
    __shared__ __align__(16) uint2    meta_i[TM * KPTS];
    __shared__ __align__(16) _Float16 As[TM * LDA];
    __shared__ __align__(16) _Float16 Bs[TN * LDA];

    const int tid  = threadIdx.x;
    const int lane = tid & 31;
    const int wave = tid >> 5;
    const int wy = wave >> 2;     // 0..1 : 32-row band
    const int wx = wave & 3;      // 0..3 : 64-col band

    const int gp0 = blockIdx.x * TM;   // global pixel-row base (within batch b)
    const int b   = gp0 >> 12;         // / HW
    const int p0  = gp0 & (HW - 1);

    const unsigned int BsAddr = (unsigned int)(unsigned long long)(&Bs[0]);
    const unsigned long long wtAddr = (unsigned long long)wt16;

    // ---- per-(row,k) bilinear metadata: 4 masked weights (x mod) + 4 corner idx
    for (int e = tid; e < TM * KPTS; e += 256) {
        const int r = e / KPTS, k = e - r * KPTS;
        const int p = p0 + r;
        const int py = p >> 6, px = p & 63;
        const float offx = offs[((size_t)b * 18 + k) * HW + p];
        const float offy = offs[((size_t)b * 18 + 9 + k) * HW + p];
        const float m    = modp[((size_t)b * 9 + k) * HW + p];
        const float sx = (float)(px + (k % 3) - 1) + offx;
        const float sy = (float)(py + (k / 3) - 1) + offy;
        const float x0 = floorf(sx), y0 = floorf(sy);
        const float fx = sx - x0,    fy = sy - y0;
        const bool vx0 = (x0 >= 0.f)  && (x0 <= 63.f);
        const bool vx1 = (x0 >= -1.f) && (x0 <= 62.f);
        const bool vy0 = (y0 >= 0.f)  && (y0 <= 63.f);
        const bool vy1 = (y0 >= -1.f) && (y0 <= 62.f);
        float4 wv;
        wv.x = (vx0 && vy0) ? (1.f - fx) * (1.f - fy) * m : 0.f;
        wv.y = (vx1 && vy0) ? fx * (1.f - fy) * m : 0.f;
        wv.z = (vx0 && vy1) ? (1.f - fx) * fy * m : 0.f;
        wv.w = (vx1 && vy1) ? fx * fy * m : 0.f;
        const int x0c = iclamp((int)x0, 0, 63);
        const int x1c = iclamp((int)x0 + 1, 0, 63);
        const int y0c = iclamp((int)y0, 0, 63);
        const int y1c = iclamp((int)y0 + 1, 0, 63);
        uint2 iv;
        iv.x = (unsigned)(y0c * 64 + x0c) | ((unsigned)(y0c * 64 + x1c) << 16);
        iv.y = (unsigned)(y1c * 64 + x0c) | ((unsigned)(y1c * 64 + x1c) << 16);
        meta_w[e] = wv;
        meta_i[e] = iv;
    }
    __syncthreads();

    v8f acc[2][4];
    #pragma unroll
    for (int rt = 0; rt < 2; ++rt)
        #pragma unroll
        for (int ct = 0; ct < 4; ++ct)
            acc[rt][ct] = (v8f){};

    const float* xb = x + (size_t)b * CIN * HW;
    const int g = lane >> 4;

    for (int chk = 0; chk < KC / TK; ++chk) {   // 72 chunks, col ordering k*256+c
        const int j0 = chk * TK;
        const int k  = j0 >> 8;       // kernel point (fixed within a chunk)
        const int c0 = j0 & 255;      // channel slab base

        // ---- B tile via Tensor Data Mover: each wave DMAs its 32-o slab.
        // src: wt16[(k*256 + wave*32) * 256 + c0], rows of 32 f16, stride 256
        // dst: Bs + wave*32*LDA halves, TDM pad gives LDS row stride 80B
        tdm_load_b_slab(BsAddr + (unsigned)(wave * 32 * LDA * 2),
                        wtAddr + 2ull * ((((unsigned long long)k * 256u + wave * 32u) << 8) + c0));

        // ---- A tile: gather-sample 64 x 32 f16 (x is L2-resident: 32MB/192MB)
        #pragma unroll 4
        for (int i = 0; i < 8; ++i) {
            const int e  = i * 256 + tid;
            const int r  = e >> 5, jj = e & 31;
            const float4 wv = meta_w[r * KPTS + k];
            const uint2  iv = meta_i[r * KPTS + k];
            const float* xp = xb + (c0 + jj) * HW;
            const float v = wv.x * xp[iv.x & 0xffffu]
                          + wv.y * xp[iv.x >> 16]
                          + wv.z * xp[iv.y & 0xffffu]
                          + wv.w * xp[iv.y >> 16];
            As[r * LDA + jj] = (_Float16)v;
        }
        // prefetch next chunk's x channel slab start lines
        if (chk + 1 < KC / TK) {
            const int c1 = ((chk + 1) * TK) & 255;
            __builtin_prefetch(xb + (c1 + (tid & 31)) * HW + ((tid >> 5) << 9), 0, 1);
        }

        __builtin_amdgcn_s_wait_tensorcnt(0);
        __syncthreads();

        // ---- fragment loads (two 16B LDS reads per fragment, per ISA layout)
        v16h af[2];
        #pragma unroll
        for (int rt = 0; rt < 2; ++rt) {
            const int r = wy * 32 + rt * 16 + (lane & 15);
            v8h lo = *(const v8h*)&As[r * LDA + g * 8];
            v8h hi = *(const v8h*)&As[r * LDA + 16 + g * 8];
            af[rt] = __builtin_shufflevector(lo, hi, 0,1,2,3,4,5,6,7,8,9,10,11,12,13,14,15);
        }
        v16h bf[4];
        #pragma unroll
        for (int ct = 0; ct < 4; ++ct) {
            const int n = wx * 64 + ct * 16 + (lane & 15);
            v8h lo = *(const v8h*)&Bs[n * LDA + g * 8];
            v8h hi = *(const v8h*)&Bs[n * LDA + 16 + g * 8];
            bf[ct] = __builtin_shufflevector(lo, hi, 0,1,2,3,4,5,6,7,8,9,10,11,12,13,14,15);
        }
        #pragma unroll
        for (int rt = 0; rt < 2; ++rt)
            #pragma unroll
            for (int ct = 0; ct < 4; ++ct)
                acc[rt][ct] = __builtin_amdgcn_wmma_f32_16x16x32_f16(
                    false, af[rt], false, bf[ct],
                    (short)0, acc[rt][ct], false, false);
        __syncthreads();
    }

    // ---- epilogue: C/D layout -> out[b, o, p] (+bias)
    const int mbase = 8 * (lane >> 4);
    const int nloc  = lane & 15;
    #pragma unroll
    for (int ct = 0; ct < 4; ++ct) {
        const int o  = wx * 64 + ct * 16 + nloc;
        const float bv = bias[o];
        float* op = out + (((size_t)b * OUTC + o) << 12);
        #pragma unroll
        for (int rt = 0; rt < 2; ++rt) {
            const int pr = p0 + wy * 32 + rt * 16 + mbase;
            #pragma unroll
            for (int j = 0; j < 8; ++j)
                op[pr + j] = acc[rt][ct][j] + bv;
        }
    }
}

// ---------------------------------------------------------------------------
extern "C" void kernel_launch(void* const* d_in, const int* in_sizes, int n_in,
                              void* d_out, int out_size, void* d_ws, size_t ws_size,
                              hipStream_t stream) {
    (void)in_sizes; (void)n_in; (void)out_size; (void)ws_size;
    const float* x  = (const float*)d_in[0];
    const float* ow = (const float*)d_in[1];
    const float* ob = (const float*)d_in[2];
    const float* mw = (const float*)d_in[3];
    const float* mb = (const float*)d_in[4];
    const float* wt = (const float*)d_in[5];
    const float* bs = (const float*)d_in[6];
    float* out  = (float*)d_out;
    float* offs = (float*)d_ws;                         // B*18*HW floats
    float* modp = offs + (size_t)NB * 18 * HW;          // B*9*HW floats
    _Float16* wt16 = (_Float16*)(modp + (size_t)NB * 9 * HW);  // 9*256*256 f16

    dcn_offmod_conv<<<(NB * 27 * HW + 255) / 256, 256, 0, stream>>>(
        x, ow, ob, mw, mb, offs, modp);
    dcn_wt_convert<<<(KPTS * OUTC * CIN + 255) / 256, 256, 0, stream>>>(wt, wt16);

    const int grid = NB * HW / TM;                      // 512
    dcn_wmma_gemm<<<grid, 256, 0, stream>>>(x, wt16, bs, offs, modp, out);
}
